// TopKRecallLoss_49838800502992
// MI455X (gfx1250) — matrix operand order repaired and checked by reading, not compiled, not run on hardware
//
#include <hip/hip_runtime.h>
#include <hip/hip_bf16.h>

typedef __attribute__((ext_vector_type(16))) _Float16 v16h;
typedef __attribute__((ext_vector_type(8)))  _Float16 v8h;
typedef __attribute__((ext_vector_type(8)))  float    v8f;

#define B_N    8192
#define D_K    128
#define TOPK   128
#define BR     16               // rows per block
#define NBLK   (B_N / BR)       // 512 blocks
#define NW     16               // waves per block (== BR, one wave per row in phase 2)
#define NT     (NW * 32)        // 512 threads
#define PITCH  8200             // halfs per sim row (8192 + 8 pad -> 16B-aligned rows)
#define HPITCH 260              // hist bins pitch: 260 mod 64 == 4 -> rows hit distinct banks

// ---- dynamic LDS layout (bytes) ----
#define OFF_SIMS   0
#define SZ_SIMS    (BR * PITCH * 2)                       // 262400
#define OFF_LAB    ((OFF_SIMS + SZ_SIMS + 15) & ~15)      // 262400
#define SZ_LAB     (B_N * 4)                              // 32768
#define OFF_HIST   ((OFF_LAB + SZ_LAB + 15) & ~15)        // 295168
#define SZ_HIST    (BR * HPITCH * 4)                      // 16640
#define OFF_SSW    ((OFF_HIST + SZ_HIST + 15) & ~15)      // 311808
#define SZ_SSW     (NW * BR * 4)                          // 1024
#define OFF_RES    ((OFF_SSW + SZ_SSW + 15) & ~15)        // 312832
#define SZ_RES     (BR * 4)
#define SMEM_BYTES (OFF_RES + SZ_RES)                     // 312896 <= 327680 (320KB WGP LDS)

// ---------------- kernel 1: L2-normalize rows, f32 -> f16 ----------------
__global__ __launch_bounds__(D_K)
void normalize_rows(const float* __restrict__ feats, _Float16* __restrict__ fnorm) {
  const int row = blockIdx.x;
  const int t   = threadIdx.x;                 // 128 threads = 4 waves
  float v  = feats[row * D_K + t];
  float ss = v * v;
  #pragma unroll
  for (int m = 16; m; m >>= 1) ss += __shfl_xor(ss, m, 32);
  __shared__ float wsum[4];
  if ((t & 31) == 0) wsum[t >> 5] = ss;
  __syncthreads();
  const float tot = wsum[0] + wsum[1] + wsum[2] + wsum[3];
  const float r   = rsqrtf(tot);
  fnorm[row * D_K + t] = (_Float16)(v * r);
}

// ---------------- kernel 2: fused WMMA sim panel + exact f16 radix top-k sum ----
__global__ __launch_bounds__(NT)
void topk_fused(const _Float16* __restrict__ fnorm,
                const int* __restrict__ labels,
                float* __restrict__ partials) {
  extern __shared__ char smem[];
  _Float16* sims = (_Float16*)(smem + OFF_SIMS);   // [BR][PITCH] f16
  int*      labS = (int*)     (smem + OFF_LAB);    // [B_N]
  unsigned* hist = (unsigned*)(smem + OFF_HIST);   // [BR][HPITCH]
  float*    ssw  = (float*)   (smem + OFF_SSW);    // [NW][BR] same-label partials
  float*    res  = (float*)   (smem + OFF_RES);    // [BR]

  const int tid     = threadIdx.x;
  const int wv      = tid >> 5;
  const int lane    = tid & 31;
  const int rowbase = blockIdx.x * BR;

  for (int i = tid; i < B_N; i += NT)           labS[i] = labels[i];
  for (int i = tid; i < BR * HPITCH; i += NT)   hist[i] = 0;
  __syncthreads();

  // -------- phase 1: 16 x 8192 similarity panel via v_wmma_f32_16x16x32_f16 ----
  const int half = lane >> 4;   // lane half: K-subgroup (A/B) / M-half (C)
  const int m16  = lane & 15;   // A: row-in-tile ; B/C: col-in-tile

  // A fragment: 16-bit A 16x32 layout (ISA 7.12.2):
  // lanes 0-15: e0-7 -> K0..7, e8-15 -> K16..23 ; lanes 16-31: +8
  v16h a[4];
  {
    const _Float16* rp = fnorm + (size_t)(rowbase + m16) * D_K;
    #pragma unroll
    for (int kk = 0; kk < 4; ++kk) {
      v8h lo = *(const v8h*)(rp + 32 * kk + 8 * half);
      v8h hi = *(const v8h*)(rp + 32 * kk + 16 + 8 * half);
      a[kk] = __builtin_shufflevector(lo, hi, 0,1,2,3,4,5,6,7,8,9,10,11,12,13,14,15);
    }
  }
  int labRow[8];
  #pragma unroll
  for (int r = 0; r < 8; ++r) labRow[r] = labS[rowbase + r + 8 * half];

  float sumsame[8];
  #pragma unroll
  for (int r = 0; r < 8; ++r) sumsame[r] = 0.f;

  for (int t = wv; t < B_N / 16; t += NW) {
    const int colbase = t * 16;
    const int col     = colbase + m16;
    const _Float16* cp = fnorm + (size_t)col * D_K;
    v8f c = {0.f, 0.f, 0.f, 0.f, 0.f, 0.f, 0.f, 0.f};
    #pragma unroll
    for (int kk = 0; kk < 4; ++kk) {
      // B fragment: 32x16, lanes 0-15: e0-7 -> K0..7, e8-15 -> K16..23 ; lanes 16-31: +8
      v8h blo = *(const v8h*)(cp + 32 * kk + 8 * half);
      v8h bhi = *(const v8h*)(cp + 32 * kk + 16 + 8 * half);
      v16h b = __builtin_shufflevector(blo, bhi, 0,1,2,3,4,5,6,7,8,9,10,11,12,13,14,15);
      c = __builtin_amdgcn_wmma_f32_16x16x32_f16(false, a[kk], false, b,
                                                 (short)0, c, false, false);
    }
    const int collab = labS[col];
    v8h hv;
    if (colbase != rowbase) {
      // fast path: no diagonal in this tile -> fully branchless epilogue
      #pragma unroll
      for (int r = 0; r < 8; ++r) {
        const float v = c[r];
        sumsame[r] += (collab == labRow[r]) ? v : 0.f;
        hv[r] = (_Float16)v;
      }
    } else {
      // one tile per block contains the diagonal (this wave only, once)
      #pragma unroll
      for (int r = 0; r < 8; ++r) {
        const int rowg = rowbase + r + 8 * half;
        const float v = c[r];
        if (rowg == col) {
          hv[r] = __builtin_bit_cast(_Float16, (unsigned short)0xFC00);  // -inf
        } else {
          hv[r] = (_Float16)v;
          sumsame[r] += (collab == labRow[r]) ? v : 0.f;
        }
      }
    }
    #pragma unroll
    for (int r = 0; r < 8; ++r)
      sims[(r + 8 * half) * PITCH + col] = hv[r];
  }

  // deterministic same-label reduction: 16 lanes of each half -> per-wave slot
  #pragma unroll
  for (int r = 0; r < 8; ++r) {
    float s = sumsame[r];
    s += __shfl_xor(s, 1, 16);
    s += __shfl_xor(s, 2, 16);
    s += __shfl_xor(s, 4, 16);
    s += __shfl_xor(s, 8, 16);
    if (m16 == 0) ssw[wv * BR + (r + 8 * half)] = s;
  }
  __syncthreads();

  // -------- phase 2: wave wv selects top-128 of row wv via 2-level radix on f16 keys ----
  {
    const int w = wv;
    unsigned* h256 = hist + w * HPITCH;
    const _Float16* srow = sims + w * PITCH;

    // pass 1: high-byte histogram of order-preserving keys, 8 halfs per ds_load_b128
    for (int i = 0; i < B_N / (32 * 8); ++i) {
      const v8h vv = *(const v8h*)(srow + i * 256 + lane * 8);
      #pragma unroll
      for (int e = 0; e < 8; ++e) {
        const unsigned bits = (unsigned)__builtin_bit_cast(unsigned short, vv[e]);
        const unsigned key  = (bits & 0x8000u) ? (bits ^ 0xFFFFu) : (bits | 0x8000u);
        atomicAdd(&h256[key >> 8], 1u);
      }
    }
    int bsel = 0, cntAbove = 0;
    if (lane == 0) {
      unsigned cnt = 0;
      for (int bin = 255; bin >= 0; --bin) {
        const unsigned c = h256[bin];
        if (cnt + c >= TOPK) { bsel = bin; break; }
        cnt += c;
      }
      cntAbove = (int)cnt;
    }
    bsel     = __shfl(bsel, 0, 32);
    cntAbove = __shfl(cntAbove, 0, 32);

    for (int i = lane; i < 256; i += 32) h256[i] = 0;   // reuse as low-byte hist

    // pass 2: sum strictly-above bins; low-byte histogram inside boundary bin
    float sumtop = 0.f;
    for (int i = 0; i < B_N / (32 * 8); ++i) {
      const v8h vv = *(const v8h*)(srow + i * 256 + lane * 8);
      #pragma unroll
      for (int e = 0; e < 8; ++e) {
        const unsigned bits = (unsigned)__builtin_bit_cast(unsigned short, vv[e]);
        const unsigned key  = (bits & 0x8000u) ? (bits ^ 0xFFFFu) : (bits | 0x8000u);
        const int hb = (int)(key >> 8);
        sumtop += (hb > bsel) ? (float)vv[e] : 0.f;
        if (hb == bsel) atomicAdd(&h256[key & 0xFFu], 1u);   // rare
      }
    }
    sumtop += __shfl_xor(sumtop, 1, 32);
    sumtop += __shfl_xor(sumtop, 2, 32);
    sumtop += __shfl_xor(sumtop, 4, 32);
    sumtop += __shfl_xor(sumtop, 8, 32);
    sumtop += __shfl_xor(sumtop, 16, 32);

    if (lane == 0) {
      const int remaining = TOPK - cntAbove;   // 1..hist[bsel]
      int cnt2 = 0; float bsum = 0.f;
      for (int l = 255; l >= 0; --l) {
        const int c = (int)h256[l];
        if (!c) continue;
        const unsigned short key2  = (unsigned short)((bsel << 8) | l);
        const unsigned short bits2 = (key2 & 0x8000u) ? (unsigned short)(key2 ^ 0x8000u)
                                                      : (unsigned short)~key2;
        const float val = (float)__builtin_bit_cast(_Float16, bits2);
        if (cnt2 + c >= remaining) { bsum += (float)(remaining - cnt2) * val; break; }
        cnt2 += c; bsum += (float)c * val;
      }
      float ssRow = 0.f;
      for (int q = 0; q < NW; ++q) ssRow += ssw[q * BR + w];
      res[w] = sumtop + bsum - ssRow;      // Σ_topk − Σ_{j≠i,same-label}
    }
  }
  __syncthreads();

  if (tid == 0) {
    float acc = 0.f;
    #pragma unroll
    for (int r = 0; r < BR; ++r) acc += res[r];
    partials[blockIdx.x] = acc;
  }
}

// ---------------- kernel 3: deterministic final reduction ----------------
__global__ __launch_bounds__(NBLK)
void finalize(const float* __restrict__ partials, float* __restrict__ out) {
  __shared__ float s[NBLK];
  const int t = threadIdx.x;
  s[t] = partials[t];
  __syncthreads();
  for (int stride = NBLK / 2; stride; stride >>= 1) {
    if (t < stride) s[t] += s[t + stride];
    __syncthreads();
  }
  if (t == 0) out[0] = s[0] / (float)B_N;
}

extern "C" void kernel_launch(void* const* d_in, const int* in_sizes, int n_in,
                              void* d_out, int out_size, void* d_ws, size_t ws_size,
                              hipStream_t stream) {
  (void)in_sizes; (void)n_in; (void)out_size; (void)ws_size;
  const float* feats  = (const float*)d_in[0];
  const int*   labels = (const int*)d_in[1];

  _Float16* fnorm    = (_Float16*)d_ws;                                   // 2 MB
  float*    partials = (float*)((char*)d_ws + (size_t)B_N * D_K * sizeof(_Float16));

  normalize_rows<<<dim3(B_N), dim3(D_K), 0, stream>>>(feats, fnorm);
  topk_fused<<<dim3(NBLK), dim3(NT), SMEM_BYTES, stream>>>(fnorm, labels, partials);
  finalize<<<dim3(1), dim3(NBLK), 0, stream>>>(partials, (float*)d_out);
}
